// GATLayer_33672543601481
// MI455X (gfx1250) — matrix operand order repaired and checked by reading, not compiled, not run on hardware
//
#include <hip/hip_runtime.h>
#include <hip/hip_bf16.h>
#include <stdint.h>

#define NB   4
#define NN   4096
#define DD   256
#define NOUT 64

typedef __attribute__((ext_vector_type(16))) __bf16 v16bf;
typedef __attribute__((ext_vector_type(8)))  float  v8f;
typedef __attribute__((ext_vector_type(4)))  float  fv4;

// Packed f32x2 -> bf16x2 (RNE) via the hardware cvt_pk path.
__device__ __forceinline__ unsigned pk2(float lo, float hi) {
  __hip_bfloat162 h = __float22bfloat162_rn(float2{lo, hi});
  union { __hip_bfloat162 h; unsigned u; } c; c.h = h;
  return c.u;                            // .x in [15:0], .y in [31:16]
}

// ---------------------------------------------------------------------------
// Fold heads + attention-src projection into the fc weights:
//   Wsb[o][d]  = sum_h W_fc[h*64+o][d]                  (h_sum weights)
//   Wsb[64][d] = sum_{h,o} Wsrc[h][o]*W_fc[h*64+o][d]   (src weights)
//   bsum[o]    = sum_h b_fc[h*64+o]
// Constant-in-j terms (b_fc·Wsrc, dst, b_attn) cancel in the softmax.
// ---------------------------------------------------------------------------
__global__ void prep_kernel(const float* __restrict__ W_fc,
                            const float* __restrict__ b_fc,
                            const float* __restrict__ W_attn,
                            float* __restrict__ Wsb, float* __restrict__ bsum) {
  int d = threadIdx.x;                  // 0..255
  float u = 0.f;
  for (int o = 0; o < 64; ++o) {
    float s = 0.f;
    for (int h = 0; h < 4; ++h) {
      float w = W_fc[(size_t)(h * 64 + o) * DD + d];
      s += w;
      float wsrc = 0.25f * (W_attn[0 * 512 + h * 128 + o] +
                            W_attn[1 * 512 + h * 128 + o] +
                            W_attn[2 * 512 + h * 128 + o] +
                            W_attn[3 * 512 + h * 128 + o]);
      u += wsrc * w;
    }
    Wsb[o * DD + d] = s;
  }
  Wsb[64 * DD + d] = u;
  if (d < 64)
    bsum[d] = b_fc[d] + b_fc[64 + d] + b_fc[128 + d] + b_fc[192 + d];
}

// ---------------------------------------------------------------------------
// Small GEMM: per node, 65 dots of length 256 -> h_sum[64] and src scalar.
// ---------------------------------------------------------------------------
__global__ __launch_bounds__(128) void fc_kernel(
    const float* __restrict__ feat, const float* __restrict__ Wsb,
    const float* __restrict__ bsum, float* __restrict__ Hsum,
    float* __restrict__ Ssrc) {
  __shared__ float f[DD];
  const int node = blockIdx.x;          // 0..B*N-1
  const int tid = threadIdx.x;          // 128
  f[tid]       = feat[(size_t)node * DD + tid];
  f[tid + 128] = feat[(size_t)node * DD + tid + 128];
  __syncthreads();
  if (tid < 65) {
    const float4* wr = (const float4*)(Wsb + (size_t)tid * DD);
    const float4* fr = (const float4*)f;
    float acc = 0.f;
#pragma unroll 8
    for (int q = 0; q < 64; ++q) {
      float4 a = wr[q], x = fr[q];
      acc += a.x * x.x + a.y * x.y + a.z * x.z + a.w * x.w;
    }
    if (tid < 64) Hsum[(size_t)node * NOUT + tid] = acc + bsum[tid];
    else          Ssrc[node] = acc;
  }
}

// ---------------------------------------------------------------------------
// Per-batch max of src (softmax stabilization).
// ---------------------------------------------------------------------------
__global__ void max_kernel(const float* __restrict__ Ssrc, float* __restrict__ maxv) {
  __shared__ float red[256];
  const int b = blockIdx.x, tid = threadIdx.x;
  float m = -3.0e38f;
  for (int j = tid; j < NN; j += 256) m = fmaxf(m, Ssrc[b * NN + j]);
  red[tid] = m; __syncthreads();
  for (int s = 128; s > 0; s >>= 1) {
    if (tid < s) red[tid] = fmaxf(red[tid], red[tid + s]);
    __syncthreads();
  }
  if (tid == 0) maxv[b] = red[0];
}

// ---------------------------------------------------------------------------
// Build G in K-PAIR-PACKED layout, ready for direct LDS staging:
//   Gp[pr][c] (dword) = { bf16(row 2pr, col c), bf16(row 2pr+1, col c) }
// cols 0..63 = e*h_sum, col 64 = e, 65..79 = 0.  pr = b*2048 + j/2.
// ---------------------------------------------------------------------------
__global__ void buildg_kernel(const float* __restrict__ Hsum,
                              const float* __restrict__ Ssrc,
                              const float* __restrict__ maxv,
                              unsigned* __restrict__ Gp) {
  const int pr = blockIdx.x * 256 + threadIdx.x;     // node pair, < B*N/2
  const int j0 = pr * 2;
  const int b  = j0 >> 12;
  const float e0 = expf(Ssrc[j0]     - maxv[b]);
  const float e1 = expf(Ssrc[j0 + 1] - maxv[b]);
  const float* h0 = Hsum + (size_t)j0 * NOUT;
  const float* h1 = h0 + NOUT;
  unsigned* g = Gp + (size_t)pr * 80;
#pragma unroll
  for (int c = 0; c < 64; ++c)
    g[c] = pk2(e0 * h0[c], e1 * h1[c]);
  g[64] = pk2(e0, e1);                               // denominator column
#pragma unroll
  for (int c = 65; c < 80; ++c) g[c] = 0u;
}

// ---------------------------------------------------------------------------
// Main aggregation GEMM per batch: C[4096 x 80] = adj(bf16) x G(bf16),
// f32 accumulate via v_wmma_f32_16x16x32_bf16. Block = 128 rows x 80 cols,
// 8 waves, each wave owns one 16-row strip x 5 N-tiles. K staged 64-deep per
// barrier (10 WMMAs per sync). adj is streamed with non-temporal loads so the
// reused G tiles stay resident in L2; G staging is a raw uint4 copy since G
// is pre-packed into K-pairs. Epilogue divides by (denominator column * N).
// ---------------------------------------------------------------------------
__global__ __launch_bounds__(256) void gat_agg_wmma(
    const float* __restrict__ adj, const unsigned* __restrict__ Gp,
    float* __restrict__ out) {
  __shared__ unsigned lds_a[128 * 33];   // 128 rows x 32 K-pairs (+1 pad)
  __shared__ unsigned lds_g[32 * 84];    // 32 K-pairs x 80 cols (stride 84)

  const int tid  = threadIdx.x;
  const int b    = blockIdx.y;
  const int row0 = blockIdx.x * 128;
  const size_t adjRow0 = (size_t)b * NN + row0;      // adj row base

  v8f acc0 = {}, acc1 = {}, acc2 = {}, acc3 = {}, acc4 = {};

  const int lane = tid & 31;
  const int w    = tid >> 5;
  const int half = lane >> 4;
  const int lrow = lane & 15;
  const int sr   = tid >> 1;             // adj staging row 0..127

  for (int k0 = 0; k0 < NN; k0 += 64) {
    __syncthreads();
    // stage adj tile: 128x64 f32 -> packed bf16 K-pairs (streaming, NT)
    {
      const float* arow = adj + (adjRow0 + sr) * (size_t)NN + k0;
#pragma unroll
      for (int cc = 0; cc < 4; cc += 2) {
        const int col = (cc + (tid & 1)) * 16;
        const fv4* src = (const fv4*)(arow + col);
        fv4 f0 = __builtin_nontemporal_load(src + 0);
        fv4 f1 = __builtin_nontemporal_load(src + 1);
        fv4 f2 = __builtin_nontemporal_load(src + 2);
        fv4 f3 = __builtin_nontemporal_load(src + 3);
        unsigned* dst = &lds_a[sr * 33 + (col >> 1)];
        dst[0] = pk2(f0.x, f0.y); dst[1] = pk2(f0.z, f0.w);
        dst[2] = pk2(f1.x, f1.y); dst[3] = pk2(f1.z, f1.w);
        dst[4] = pk2(f2.x, f2.y); dst[5] = pk2(f2.z, f2.w);
        dst[6] = pk2(f3.x, f3.y); dst[7] = pk2(f3.z, f3.w);
      }
      if (k0 + 64 < NN)                  // pull next tile toward the WGP
        __builtin_prefetch(arow + 64 + (tid & 1) * 32, 0, 1);
    }
    // stage G tile: 32 K-pair rows x 80 dwords, contiguous in global ->
    // plain uint4 copy (LDS stride 84 dwords to avoid bank conflicts)
    {
      const uint4* gp4 =
          (const uint4*)(Gp + ((size_t)b * (NN / 2) + (k0 >> 1)) * 80);
      for (int idx = tid; idx < 640; idx += 256) {
        int p  = idx / 20;               // K-pair row within tile
        int c4 = idx - p * 20;           // uint4 column
        *(uint4*)&lds_g[p * 84 + c4 * 4] = gp4[idx];
      }
    }
    __syncthreads();

#pragma unroll
    for (int s = 0; s < 2; ++s) {        // two K=32 WMMA substeps
      // A fragment (16x32 bf16): lanes 0-15 K-pairs {0..7},
      // lanes 16-31 K-pairs {8..15}
      union { v16bf v; unsigned u[8]; } af;
      const unsigned* ar = &lds_a[(w * 16 + lrow) * 33 + s * 16 + half * 8];
      af.u[0] = ar[0];  af.u[1] = ar[1];  af.u[2] = ar[2];  af.u[3] = ar[3];
      af.u[4] = ar[4];  af.u[5] = ar[5];  af.u[6] = ar[6];  af.u[7] = ar[7];

      // B fragments (32x16 bf16): lanes 0-15 K 0..15, lanes 16-31 K 16..31
      const unsigned* gb = &lds_g[(s * 16 + half * 8) * 84 + lrow];
#define BFRAG(T, ACC)                                                         \
      {                                                                       \
        union { v16bf v; unsigned u[8]; } bfm;                                \
        const unsigned* br = gb + (T) * 16;                                   \
        bfm.u[0] = br[0];       bfm.u[1] = br[84];                            \
        bfm.u[2] = br[2 * 84];  bfm.u[3] = br[3 * 84];                        \
        bfm.u[4] = br[4 * 84];  bfm.u[5] = br[5 * 84];                        \
        bfm.u[6] = br[6 * 84];  bfm.u[7] = br[7 * 84];                        \
        ACC = __builtin_amdgcn_wmma_f32_16x16x32_bf16(                        \
            false, af.v, false, bfm.v, (short)0, ACC, false, false);          \
      }
      BFRAG(0, acc0) BFRAG(1, acc1) BFRAG(2, acc2) BFRAG(3, acc3) BFRAG(4, acc4)
#undef BFRAG
    }
  }

  // Epilogue: denom lives in tile-4 local col 0 -> lane 0 (rows M=v) /
  // lane 16 (rows M=v+8) per the 16x16 f32 C layout.
  const float scale = 1.0f / (float)NN;
#pragma unroll
  for (int v = 0; v < 8; ++v) {
    float dv  = __shfl(acc4[v], lane & 16, 32);
    float inv = scale / dv;
    int m = row0 + w * 16 + v + half * 8;
    float* orow = out + ((size_t)b * NN + m) * NOUT + lrow;
    orow[0]  = acc0[v] * inv;
    orow[16] = acc1[v] * inv;
    orow[32] = acc2[v] * inv;
    orow[48] = acc3[v] * inv;
  }
}

// ---------------------------------------------------------------------------
extern "C" void kernel_launch(void* const* d_in, const int* in_sizes, int n_in,
                              void* d_out, int out_size, void* d_ws, size_t ws_size,
                              hipStream_t stream) {
  const float* features = (const float*)d_in[0];
  const float* adj      = (const float*)d_in[1];
  const float* W_fc     = (const float*)d_in[2];
  const float* b_fc     = (const float*)d_in[3];
  const float* W_attn   = (const float*)d_in[4];
  // d_in[5] (b_attn) is constant over j -> cancels in softmax; unused.
  (void)in_sizes; (void)n_in; (void)out_size; (void)ws_size;

  char* ws = (char*)d_ws;                 // ~6.7 MB used
  float*    Wsb  = (float*)(ws + 0);          // 65*256 f32   = 66560 B
  float*    bsum = (float*)(ws + 66560);      // 64 f32
  float*    Ssrc = (float*)(ws + 66816);      // B*N f32      = 65536 B
  float*    maxv = (float*)(ws + 132352);     // B f32
  float*    Hsum = (float*)(ws + 132368);     // B*N*64 f32   = 4 MB
  unsigned* Gp   = (unsigned*)(ws + 4326672); // B*N/2*80 dw  = 2.5 MB
  float* out = (float*)d_out;

  prep_kernel  <<<dim3(1),               dim3(256), 0, stream>>>(W_fc, b_fc, W_attn, Wsb, bsum);
  fc_kernel    <<<dim3(NB * NN),         dim3(128), 0, stream>>>(features, Wsb, bsum, Hsum, Ssrc);
  max_kernel   <<<dim3(NB),              dim3(256), 0, stream>>>(Ssrc, maxv);
  buildg_kernel<<<dim3(NB * NN / 512),   dim3(256), 0, stream>>>(Hsum, Ssrc, maxv, Gp);
  gat_agg_wmma <<<dim3(NN / 128, NB),    dim3(256), 0, stream>>>(adj, Gp, out);
}
